// Self_Align_attention_8358006358494
// MI455X (gfx1250) — matrix operand order repaired and checked by reading, not compiled, not run on hardware
//
#include <hip/hip_runtime.h>
#include <stdint.h>

// ---------------------------------------------------------------------------
// Fused self-align attention + SFU gating for MI455X (gfx1250, wave32, WMMA).
//
// Prep kernels:  Wr/Wg [1024,256] f32 -> transposed bf16 [256,1024] in ws.
//                c [B,JX,D] f32 -> bf16 cb [B,JX,D] and transposed ct [B,D,JX].
// Fused kernel:  per block = 16 query rows of one batch:
//   - flash-attention over 64 key tiles of 32 (bf16 WMMA, f32 online softmax)
//     with GLOBAL_LOAD_ASYNC_TO_LDS_B128 tile staging (ASYNCcnt)
//   - gating GEMM cat[16,1024] x Wt[1024,256] (bf16 WMMA) + tanh/sigmoid blend
// ---------------------------------------------------------------------------

typedef __bf16 bf16_t;
typedef __attribute__((ext_vector_type(16))) __bf16 v16bf;
typedef __attribute__((ext_vector_type(8)))  float  v8f;
typedef int v4i __attribute__((vector_size(16)));

#define AS1 __attribute__((address_space(1)))
#define AS3 __attribute__((address_space(3)))

#if defined(__has_builtin)
#if __has_builtin(__builtin_amdgcn_global_load_async_to_lds_b128)
#define HAVE_ASYNC_LDS 1
#endif
#if __has_builtin(__builtin_amdgcn_s_wait_asynccnt)
#define HAVE_WAIT_ASYNC 1
#endif
#endif

__device__ __forceinline__ void wait_asynccnt0() {
#if defined(HAVE_WAIT_ASYNC)
  __builtin_amdgcn_s_wait_asynccnt(0);
#else
  asm volatile("s_wait_asynccnt 0" ::: "memory");
#endif
}

#define WMMA_BF16(C, A, Bv)                                                    \
  (C) = __builtin_amdgcn_wmma_f32_16x16x32_bf16(false, (A), false, (Bv),       \
                                                (short)0, (C), false, false)

static constexpr int BATCH = 8;
static constexpr int JX    = 2048;
static constexpr int D     = 256;
static constexpr int KTILE = 32;             // keys per flash iteration
static constexpr int NKT   = JX / KTILE;     // 64 key tiles
static constexpr int RS    = 272;            // krow stride (halves): 544B, 32B-aligned
static constexpr int KS    = 48;             // kcol stride (halves): 96B
static constexpr int PS    = 48;             // pstage row stride (halves)
static constexpr int FS    = 1040;           // f-tile row stride (halves): 2080B

union Frag16 { v16bf v; uint4 q[2]; };

__device__ __forceinline__ v16bf frag_contig(const bf16_t* p) {
  Frag16 u;
  u.q[0] = *(const uint4*)(p);
  u.q[1] = *(const uint4*)(p + 8);
  return u.v;
}
__device__ __forceinline__ v16bf frag_split(const bf16_t* p0, const bf16_t* p1) {
  Frag16 u;
  u.q[0] = *(const uint4*)(p0);
  u.q[1] = *(const uint4*)(p1);
  return u.v;
}

// reductions across the 16 lanes of a half-wave (rows of a C-layout tile)
__device__ __forceinline__ float half_max(float x) {
#pragma unroll
  for (int off = 8; off > 0; off >>= 1) x = fmaxf(x, __shfl_xor(x, off, 32));
  return x;
}
__device__ __forceinline__ float half_sum(float x) {
#pragma unroll
  for (int off = 8; off > 0; off >>= 1) x += __shfl_xor(x, off, 32);
  return x;
}

struct SharedAttn {
  bf16_t krow[KTILE * RS];        // key tile row-major   [key][d]   17408 B
  bf16_t kcol[D * KS];            // key tile col-major   [d][key]   24576 B
  bf16_t pstage[8 * 16 * PS];     // per-wave P staging              12288 B
};
struct SharedGemm {
  float  qtile[16 * D];           // q_a tile f32                    16384 B
  bf16_t f[16 * FS];              // cat tile bf16                   33280 B
};
union SharedMem { SharedAttn a; SharedGemm g; };

__global__ void prep_weights_kernel(const float* __restrict__ Wr,
                                    const float* __restrict__ Wg,
                                    bf16_t* __restrict__ WtR,
                                    bf16_t* __restrict__ WtG) {
  int o = blockIdx.x * 256 + threadIdx.x;      // o = col*1024 + fi
  int col = o >> 10;
  int fi  = o & 1023;
  WtR[o] = (bf16_t)Wr[fi * D + col];
  WtG[o] = (bf16_t)Wg[fi * D + col];
}

__global__ void prep_c_kernel(const float* __restrict__ c,
                              bf16_t* __restrict__ cb,
                              bf16_t* __restrict__ ct) {
  size_t idx = (size_t)blockIdx.x * 256 + threadIdx.x;   // over B*JX*D
  int col = (int)(idx & (D - 1));
  int row = (int)((idx >> 8) & (JX - 1));
  int b   = (int)(idx >> 19);
  bf16_t v = (bf16_t)c[idx];
  cb[idx] = v;
  ct[((size_t)(b * D + col)) * JX + row] = v;
}

__global__ __launch_bounds__(256)
void fused_attn_gate_kernel(const float* __restrict__ c,
                            const bf16_t* __restrict__ cb,
                            const bf16_t* __restrict__ ct,
                            const uint8_t* __restrict__ mask,
                            const bf16_t* __restrict__ WtR,
                            const bf16_t* __restrict__ WtG,
                            const float* __restrict__ Br,
                            const float* __restrict__ Bg,
                            float* __restrict__ out) {
  __shared__ SharedMem smem;

  const int tid   = threadIdx.x;
  const int wave  = tid >> 5;
  const int lane  = tid & 31;
  const int hw    = lane >> 4;        // half-wave: 0 or 1
  const int ln    = lane & 15;
  const int wcol  = wave * 32;        // this wave's 32-column slice of d

  const int blk   = blockIdx.x;
  const int b     = blk >> 7;         // 128 row-tiles per batch
  const int qbase = (blk & 127) << 4; // 16 query rows

  // ---- Q A-fragments: 16 rows x 256 d, bf16, kept in VGPRs (8 chunks) ----
  const bf16_t* qrow = cb + ((size_t)(b * JX + qbase + ln)) * D;
  v16bf qa[8];
#pragma unroll
  for (int kc = 0; kc < 8; ++kc) {
    const bf16_t* p = qrow + kc * 32 + hw * 16;
    qa[kc] = frag_split(p, p + 8);
  }

  // query mask bits for the rows this lane's C-layout slots cover
  int qmb = 0;
#pragma unroll
  for (int v = 0; v < 8; ++v)
    qmb |= (mask[b * JX + qbase + v + 8 * hw] ? 1 : 0) << v;

  v8f o0 = {0.f,0.f,0.f,0.f,0.f,0.f,0.f,0.f};
  v8f o1 = o0;
  float m[8], l[8];
#pragma unroll
  for (int v = 0; v < 8; ++v) { m[v] = -INFINITY; l[v] = 0.f; }

  // ------------------------- flash-attention loop -------------------------
  for (int kt = 0; kt < NKT; ++kt) {
    const int kbase = kt * KTILE;
    __syncthreads();
    // stage key tile (row-major from cb, col-major from ct) into LDS
    {
      const bf16_t* cbb = cb + ((size_t)(b * JX + kbase)) * D;
      const bf16_t* ctb = ct + (size_t)b * D * JX + kbase;
#if defined(HAVE_ASYNC_LDS)
#pragma unroll
      for (int t = 0; t < 4; ++t) {            // krow: 32 rows x 32 chunks
        int id = t * 256 + tid;
        int row = id >> 5, ch = id & 31;
        __builtin_amdgcn_global_load_async_to_lds_b128(
            (AS1 v4i*)(cbb + row * D + ch * 8),
            (AS3 v4i*)(&smem.a.krow[row * RS + ch * 8]), 0, 0);
      }
#pragma unroll
      for (int t = 0; t < 4; ++t) {            // kcol: 256 cols x 4 chunks
        int id = t * 256 + tid;
        int col = id >> 2, ch = id & 3;
        __builtin_amdgcn_global_load_async_to_lds_b128(
            (AS1 v4i*)(ctb + (size_t)col * JX + ch * 8),
            (AS3 v4i*)(&smem.a.kcol[col * KS + ch * 8]), 0, 0);
      }
      wait_asynccnt0();
#else
      for (int i = 0; i < 32; ++i) {           // fallback: copy via VGPRs
        int idx = i * 256 + tid;
        int row = idx >> 8, col = idx & 255;
        smem.a.krow[row * RS + col] = cbb[row * D + col];
      }
      for (int i = 0; i < 32; ++i) {
        int idx = i * 256 + tid;
        int col = idx >> 5, k = idx & 31;
        smem.a.kcol[col * KS + k] = ctb[(size_t)col * JX + k];
      }
#endif
    }
    __syncthreads();

    // S = Q @ K^T : two 16x16 tiles (keys ln and ln+16)
    v8f s0 = {0.f,0.f,0.f,0.f,0.f,0.f,0.f,0.f};
    v8f s1 = s0;
#pragma unroll
    for (int kc = 0; kc < 8; ++kc) {
      v16bf bk0 = frag_contig(&smem.a.krow[ln        * RS + kc * 32 + hw * 16]);
      v16bf bk1 = frag_contig(&smem.a.krow[(ln + 16) * RS + kc * 32 + hw * 16]);
      WMMA_BF16(s0, qa[kc], bk0);
      WMMA_BF16(s1, qa[kc], bk1);
    }

    // diagonal-zero + pair mask, then online softmax update
    const int ki0 = kbase + ln, ki1 = kbase + ln + 16;
    const bool km0 = mask[b * JX + ki0] != 0;
    const bool km1 = mask[b * JX + ki1] != 0;
#pragma unroll
    for (int v = 0; v < 8; ++v) {
      const int qi = qbase + v + 8 * hw;
      const bool qm = (qmb >> v) & 1;
      float x0 = s0[v], x1 = s1[v];
      if (qi == ki0) x0 = 0.f;
      if (qi == ki1) x1 = 0.f;
      if (!(qm && km0)) x0 += -1e30f;
      if (!(qm && km1)) x1 += -1e30f;
      float rmax = half_max(fmaxf(x0, x1));
      float mnew = fmaxf(m[v], rmax);
      float sc   = __expf(m[v] - mnew);   // expf(-inf)=0 on first tile
      m[v] = mnew;
      float p0 = __expf(x0 - mnew);
      float p1 = __expf(x1 - mnew);
      s0[v] = p0; s1[v] = p1;
      l[v]  = l[v] * sc + half_sum(p0 + p1);
      o0[v] *= sc;
      o1[v] *= sc;
    }

    // stage P (C-layout -> A-layout) through wave-private LDS
    bf16_t* pw = &smem.a.pstage[wave * 16 * PS];
#pragma unroll
    for (int v = 0; v < 8; ++v) {
      const int row = v + 8 * hw;
      pw[row * PS + ln]      = (bf16_t)s0[v];
      pw[row * PS + ln + 16] = (bf16_t)s1[v];
    }
    asm volatile("s_wait_dscnt 0" ::: "memory");
    const bf16_t* pp = pw + ln * PS + hw * 16;
    v16bf pa = frag_split(pp, pp + 8);

    // O += P @ V (this wave's two 16-column tiles of d)
    v16bf vb0 = frag_contig(&smem.a.kcol[(wcol + ln)      * KS + hw * 16]);
    v16bf vb1 = frag_contig(&smem.a.kcol[(wcol + 16 + ln) * KS + hw * 16]);
    WMMA_BF16(o0, pa, vb0);
    WMMA_BF16(o1, pa, vb1);
  }

  // ---------------- attention epilogue: O /= l -> qtile (LDS) -------------
#pragma unroll
  for (int v = 0; v < 8; ++v) {
    float inv = 1.0f / l[v];
    o0[v] *= inv;
    o1[v] *= inv;
  }
  __syncthreads();                 // everyone done with attn LDS before reuse
  float* qt = smem.g.qtile;
#pragma unroll
  for (int v = 0; v < 8; ++v) {
    const int row = v + 8 * hw;
    qt[row * D + wcol + ln]      = o0[v];
    qt[row * D + wcol + 16 + ln] = o1[v];
  }
  __syncthreads();

  // ---------------- build cat = [c, q, c*q, c-q] in bf16 ------------------
  for (int i = 0; i < 64; ++i) {
    int idx = i * 256 + tid;
    int row = idx >> 10, fi = idx & 1023;
    int seg = fi >> 8,   j  = fi & 255;
    float qv = qt[row * D + j];
    float cv = c[((size_t)(b * JX + qbase + row)) * D + j];
    float val = (seg == 0) ? cv : (seg == 1) ? qv : (seg == 2) ? cv * qv
                                                              : (cv - qv);
    smem.g.f[row * FS + fi] = (bf16_t)val;
  }
  __syncthreads();

  // ---------------- gating GEMM: cat @ WtR^T / WtG^T ----------------------
  v8f r0 = {0.f,0.f,0.f,0.f,0.f,0.f,0.f,0.f};
  v8f r1 = r0, g0 = r0, g1 = r0;
  const bf16_t* fb = smem.g.f + ln * FS;
  for (int kc = 0; kc < 32; ++kc) {
    const bf16_t* ap = fb + kc * 32 + hw * 16;
    v16bf a = frag_split(ap, ap + 8);
    const int fi = kc * 32 + hw * 16;
    v16bf br0 = frag_contig(WtR + (size_t)(wcol + ln)      * 1024 + fi);
    v16bf br1 = frag_contig(WtR + (size_t)(wcol + 16 + ln) * 1024 + fi);
    v16bf bg0 = frag_contig(WtG + (size_t)(wcol + ln)      * 1024 + fi);
    v16bf bg1 = frag_contig(WtG + (size_t)(wcol + 16 + ln) * 1024 + fi);
    WMMA_BF16(r0, a, br0);
    WMMA_BF16(r1, a, br1);
    WMMA_BF16(g0, a, bg0);
    WMMA_BF16(g1, a, bg1);
  }

  // ---------------- epilogue: out = g*tanh(r) + (1-g)*c -------------------
#pragma unroll
  for (int v = 0; v < 8; ++v) {
    const int row  = qbase + v + 8 * hw;
    const size_t brow = (size_t)(b * JX + row);
#pragma unroll
    for (int t = 0; t < 2; ++t) {
      const int col = wcol + 16 * t + ln;
      float ar = t ? r1[v] : r0[v];
      float ag = t ? g1[v] : g0[v];
      float rr = tanhf(ar + Br[col]);
      float gg = 1.0f / (1.0f + __expf(-(ag + Bg[col])));
      float cv = c[brow * D + col];
      out[brow * D + col] = gg * rr + (1.0f - gg) * cv;
    }
  }
}

extern "C" void kernel_launch(void* const* d_in, const int* in_sizes, int n_in,
                              void* d_out, int out_size, void* d_ws, size_t ws_size,
                              hipStream_t stream) {
  (void)in_sizes; (void)n_in; (void)out_size; (void)ws_size;
  const float*   c    = (const float*)d_in[0];
  const uint8_t* mask = (const uint8_t*)d_in[1];
  const float*   Wr   = (const float*)d_in[2];
  const float*   Br   = (const float*)d_in[3];
  const float*   Wg   = (const float*)d_in[4];
  const float*   Bg   = (const float*)d_in[5];
  float*         out  = (float*)d_out;

  bf16_t* WtR = (bf16_t*)d_ws;                       // [256][1024] bf16
  bf16_t* WtG = WtR + (size_t)D * 1024;              // [256][1024] bf16
  bf16_t* cb  = WtG + (size_t)D * 1024;              // [B,JX,D]  bf16
  bf16_t* ct  = cb  + (size_t)BATCH * JX * D;        // [B,D,JX]  bf16

  prep_weights_kernel<<<(D * 1024) / 256, 256, 0, stream>>>(Wr, Wg, WtR, WtG);
  prep_c_kernel<<<(BATCH * JX * D) / 256, 256, 0, stream>>>(c, cb, ct);

  const int grid = BATCH * (JX / 16);                // 1024 blocks
  fused_attn_gate_kernel<<<grid, 256, 0, stream>>>(c, cb, ct, mask, WtR, WtG,
                                                   Br, Bg, out);
}
